// MultiHeadAttention_74792560493185
// MI455X (gfx1250) — matrix operand order repaired and checked by reading, not compiled, not run on hardware
//
#include <hip/hip_runtime.h>

// ---------------------------------------------------------------------------
// Multi-head attention forward for MI455X (gfx1250).
// bf16 WMMA pipeline, LDS-tiled GEMMs (4 waves / 64x64 tile), flash attention
// with double-buffered TDM (tensor_load_to_lds) K/V staging.
// B=2, S=2048, D=1024, H=16, dh=64.
// ---------------------------------------------------------------------------

typedef __attribute__((ext_vector_type(16))) __bf16    v16bf;
typedef __attribute__((ext_vector_type(8)))  float     v8f;
typedef __attribute__((ext_vector_type(8)))  unsigned  v8u;
typedef __attribute__((ext_vector_type(4)))  unsigned  uint32x4;
typedef __attribute__((ext_vector_type(8)))  int       int32x8;
typedef __attribute__((ext_vector_type(4)))  int       int32x4;

#define WMMA_BF16(a, b, c) \
  __builtin_amdgcn_wmma_f32_16x16x32_bf16(false, (a), false, (b), (short)0, (c), false, false)

#if defined(__has_builtin)
#if __has_builtin(__builtin_amdgcn_tensor_load_to_lds) && \
    __has_builtin(__builtin_amdgcn_s_wait_tensorcnt)
#define HAVE_TDM 1
#endif
#endif
#if __has_include(<hip/amd_detail/amd_gfx1250_TDM.h>)
#define TDM_SIX_ARGS 1
#endif

namespace {
constexpr int BB = 2;      // batch
constexpr int S  = 2048;   // sequence
constexpr int D  = 1024;   // model dim
constexpr int H  = 16;     // heads
constexpr int DH = 64;     // head dim
// 1/sqrt(64) * log2(e): logits land in base-2 domain so softmax is exp2f.
constexpr float QSCALE = 0.125f * 1.44269504088896340736f;
}

__device__ __forceinline__ unsigned short bf16bits(float x) {
  return __builtin_bit_cast(unsigned short, (__bf16)x);
}
__device__ __forceinline__ unsigned pack_bf16(float lo, float hi) {
  return (unsigned)bf16bits(lo) | ((unsigned)bf16bits(hi) << 16);
}

#ifdef HAVE_TDM
// 2D tensor-DMA load into LDS. Dimensions in 4-byte elements (data_size=4B).
// D# layout per CDNA5 ISA ch.8: group0 = {flags, lds_addr, global_addr, type},
// group1 = {mask/dsize, tensor dims, tile dims, dim0 stride}.
__device__ __forceinline__ void tdm_load_2d(unsigned lds_byte_off,
                                            const void* gaddr,
                                            unsigned tile_d0, unsigned tile_d1,
                                            unsigned tens_d0, unsigned tens_d1,
                                            unsigned stride0_elems) {
  const unsigned long long ga = (unsigned long long)(size_t)gaddr;
  uint32x4 g0;
  g0[0] = 1u;                                   // count=1, user descriptor
  g0[1] = lds_byte_off;                         // lds_addr
  g0[2] = (unsigned)ga;                         // global_addr[31:0]
  g0[3] = (unsigned)((ga >> 32) & 0x01FFFFFFu)  // global_addr[56:32]
          | 0x80000000u;                        // type = 2 ("image")
  int32x8 g1;
  g1[0] = (int)(2u << 16);                      // data_size = 4 bytes
  g1[1] = (int)((tens_d0 & 0xFFFFu) << 16);     // tensor_dim0[15:0]
  g1[2] = (int)((tens_d0 >> 16) | ((tens_d1 & 0xFFFFu) << 16));
  g1[3] = (int)((tens_d1 >> 16) | (tile_d0 << 16));
  g1[4] = (int)(tile_d1 & 0xFFFFu);             // tile_dim1 (tile_dim2=0)
  g1[5] = (int)stride0_elems;                   // tensor_dim0_stride[31:0]
  g1[6] = 0;
  g1[7] = 0;
  const int32x4 z4 = {0, 0, 0, 0};
#ifdef TDM_SIX_ARGS
  const int32x8 z8 = {0, 0, 0, 0, 0, 0, 0, 0};
  __builtin_amdgcn_tensor_load_to_lds(g0, g1, z4, z4, z8, 0);
#else
  __builtin_amdgcn_tensor_load_to_lds(g0, g1, z4, z4, 0);
#endif
}
#endif  // HAVE_TDM

// ---------------------------------------------------------------------------
// QKV projection: Y = X @ W + bias, 64x64 tile per 128-thread block.
// LDS-staged bf16 tiles; each wave computes 16 rows x 64 cols (4 WMMA/k-step).
//   mode 0 (Q): bf16 [bh][s][dh], pre-scaled by QSCALE   (A-frag friendly)
//   mode 1 (K): u32  [bh][dh/2][s]  (dh-pair packed)      (B-frag for QK^T)
//   mode 2 (V): u32  [bh][s/2][dh]  (key-pair packed)     (B-frag for PV)
// ---------------------------------------------------------------------------
__global__ __launch_bounds__(128)
void qkv_proj_kernel(const float* __restrict__ X, const float* __restrict__ W,
                     const float* __restrict__ bias,
                     unsigned short* __restrict__ outQ,
                     unsigned* __restrict__ outKV, int mode) {
  __shared__ unsigned lAu[64 * 16];   // A tile: 64 rows x 32 k  (bf16 pairs)
  __shared__ unsigned lBu[16 * 64];   // B tile: 16 kp  x 64 col (bf16 pairs)

  const int tid  = threadIdx.x;
  const int wave = tid >> 5;
  const int lane = tid & 31;
  const int half = lane >> 4;
  const int lr   = lane & 15;
  const int r0 = blockIdx.x * 64;     // rows of X (b*S+s)
  const int c0 = blockIdx.y * 64;     // cols of W (h*64+dh)

  v8f acc0 = {}, acc1 = {}, acc2 = {}, acc3 = {};

  for (int k = 0; k < D; k += 32) {
    __syncthreads();
    // ---- cooperative A tile: thread copies 16 floats of one row, cvt->bf16
    {
      const int ar = tid >> 1;
      const int kc = (tid & 1) * 16;                   // float offset in slice
      const float4* src = (const float4*)(X + (size_t)(r0 + ar) * D + k + kc);
#pragma unroll
      for (int i = 0; i < 4; ++i) {
        const float4 f = src[i];
        lAu[ar * 16 + (kc >> 1) + 2 * i]     = pack_bf16(f.x, f.y);
        lAu[ar * 16 + (kc >> 1) + 2 * i + 1] = pack_bf16(f.z, f.w);
      }
    }
    // ---- cooperative B tile: pack k-pairs (rows 2kp, 2kp+1) per column
    {
      const int kp = tid >> 3;                         // 0..15
      const int cc = (tid & 7) * 8;                    // 0..56
      const float* rp0 = W + (size_t)(k + 2 * kp) * D + c0 + cc;
      const float* rp1 = rp0 + D;
#pragma unroll
      for (int i = 0; i < 8; ++i)
        lBu[kp * 64 + cc + i] = pack_bf16(rp0[i], rp1[i]);
      if (k + 32 < D)   // prefetch next k-slice of W (global_prefetch_b8)
        __builtin_prefetch(W + (size_t)(k + 32 + 2 * kp) * D + c0 + cc, 0, 1);
    }
    __syncthreads();

    // ---- A fragment for this wave's 16 rows
    v16bf a;
    {
      v8u u;
#pragma unroll
      for (int v = 0; v < 8; ++v) {
        const int kk = 16 * half + 2 * v;
        u[v] = lAu[(wave * 16 + lr) * 16 + (kk >> 1)];
      }
      a = __builtin_bit_cast(v16bf, u);
    }
    // ---- 4 column tiles
#pragma unroll
    for (int t = 0; t < 4; ++t) {
      v8u u;
#pragma unroll
      for (int v = 0; v < 8; ++v)
        u[v] = lBu[(8 * half + v) * 64 + t * 16 + lr];
      v8f& acc = (t == 0 ? acc0 : t == 1 ? acc1 : t == 2 ? acc2 : acc3);
      acc = WMMA_BF16(a, __builtin_bit_cast(v16bf, u), acc);
    }
  }

  // ---- epilogue: mode-specific packed writes
  const int row0 = r0 + wave * 16;
#pragma unroll
  for (int t = 0; t < 4; ++t) {
    const v8f acc = (t == 0 ? acc0 : t == 1 ? acc1 : t == 2 ? acc2 : acc3);
    const int colB = c0 + t * 16 + lr;
    const float bb = bias[colB];
    const int h  = colB >> 6;
    const int dh = colB & 63;

    if (mode == 0) {
#pragma unroll
      for (int v = 0; v < 8; ++v) {
        const int r = row0 + v + 8 * half;            // C layout: M = v+8*half
        const int bidx = r / S, s = r % S;
        const float val = (acc[v] + bb) * QSCALE;
        outQ[((size_t)(bidx * H + h) * S + s) * DH + dh] = bf16bits(val);
      }
    } else if (mode == 1) {
#pragma unroll
      for (int v = 0; v < 8; ++v) {
        const float val   = acc[v] + bb;
        const float other = __shfl_xor(val, 1, 32);   // neighbor dh, same row
        if ((lane & 1) == 0) {
          const int r = row0 + v + 8 * half;
          const int bidx = r / S, s = r % S;
          outKV[((size_t)(bidx * H + h) * (DH / 2) + (dh >> 1)) * S + s] =
              pack_bf16(val, other);
        }
      }
    } else {
#pragma unroll
      for (int i = 0; i < 4; ++i) {                   // key pairs: rows 2i,2i+1
        const float v0 = acc[2 * i] + bb;
        const float v1 = acc[2 * i + 1] + bb;
        const int r = row0 + 2 * i + 8 * half;
        const int bidx = r / S, s = r % S;
        outKV[((size_t)(bidx * H + h) * (S / 2) + (s >> 1)) * DH + dh] =
            pack_bf16(v0, v1);
      }
    }
  }
}

// ---------------------------------------------------------------------------
// Flash attention: 128 threads = 4 waves, 64 query rows per block.
// K/V tiles (32 keys) double-buffered in LDS via TDM: tile j+1 is DMA'd while
// tile j feeds 8 WMMAs + softmax; s_wait_tensorcnt<=2 retires the consumed
// pair (TENSORcnt is in-order). Fallback: plain cooperative copies.
// ---------------------------------------------------------------------------
__global__ __launch_bounds__(128)
void attn_kernel(const unsigned* __restrict__ Qp,    // [bh][s][32] u32
                 const unsigned* __restrict__ Kp,    // [bh][32][S] u32
                 const unsigned* __restrict__ Vp,    // [bh][S/2][64] u32
                 unsigned short* __restrict__ attnOut) {  // bf16 [4096][1024]
  __shared__ unsigned lK[2][32 * 32];       // [buf][d2][key]   2 x 4 KB
  __shared__ unsigned lV[2][16 * 64];       // [buf][kpair][dh] 2 x 4 KB
  __shared__ unsigned short lP[4][16 * 32]; // per-wave P tile, 4 KB

  const int tid  = threadIdx.x;
  const int wave = tid >> 5;
  const int lane = tid & 31;
  const int half = lane >> 4;
  const int lr   = lane & 15;
  const int q0 = blockIdx.x * 64 + wave * 16;
  const int h  = blockIdx.y;
  const int b  = blockIdx.z;
  const int bh = b * H + h;

  // Q fragments: 16x64 -> two 16x32 A fragments (dh 0..31, 32..63)
  v16bf aq0, aq1;
  {
    v8u u0, u1;
#pragma unroll
    for (int v = 0; v < 8; ++v) {
      const int d2 = 8 * half + v;
      const size_t base = ((size_t)bh * S + (q0 + lr)) * (DH / 2);
      u0[v] = Qp[base + d2];
      u1[v] = Qp[base + d2 + 16];
    }
    aq0 = __builtin_bit_cast(v16bf, u0);
    aq1 = __builtin_bit_cast(v16bf, u1);
  }

  v8f o0 = {}, o1 = {}, o2 = {}, o3 = {};
  float m[8], ssum[8];
#pragma unroll
  for (int v = 0; v < 8; ++v) { m[v] = -1e30f; ssum[v] = 0.0f; }

#ifdef HAVE_TDM
  // Prologue: DMA tile 0 into buffer 0.
  if (wave == 0) {
    tdm_load_2d((unsigned)(size_t)(void*)&lK[0][0],
                Kp + (size_t)bh * (DH / 2) * S,
                /*tile*/ 32, 32, /*tensor*/ S, DH / 2, /*stride*/ S);
    tdm_load_2d((unsigned)(size_t)(void*)&lV[0][0],
                Vp + (size_t)bh * (S / 2) * DH,
                /*tile*/ 64, 16, /*tensor*/ DH, S / 2, /*stride*/ DH);
  }
#endif

  for (int j = 0; j < S / 32; ++j) {
    const int buf = j & 1;
    __syncthreads();   // all waves done reading the buffer we (over)write next

#ifdef HAVE_TDM
    if (wave == 0) {
      if (j + 1 < S / 32) {
        const int k1 = (j + 1) * 32;
        tdm_load_2d((unsigned)(size_t)(void*)&lK[1 - buf][0],
                    Kp + (size_t)bh * (DH / 2) * S + k1,
                    /*tile*/ 32, 32, /*tensor*/ S, DH / 2, /*stride*/ S);
        tdm_load_2d((unsigned)(size_t)(void*)&lV[1 - buf][0],
                    Vp + ((size_t)bh * (S / 2) + (j + 1) * 16) * DH,
                    /*tile*/ 64, 16, /*tensor*/ DH, S / 2, /*stride*/ DH);
        __builtin_amdgcn_s_wait_tensorcnt(2);  // retire tile j's pair
      } else {
        __builtin_amdgcn_s_wait_tensorcnt(0);
      }
    }
#else
    {
      const int k0 = j * 32;
      // K: row d2 = tid>>2, 8 consecutive keys per thread
      const uint4* ks = (const uint4*)(Kp + ((size_t)bh * (DH / 2) + (tid >> 2)) * S
                                       + k0 + (tid & 3) * 8);
      uint4* dK = (uint4*)&lK[buf][0];
      dK[tid * 2]     = ks[0];
      dK[tid * 2 + 1] = ks[1];
      // V: fully contiguous 4 KB
      const uint4* vs = (const uint4*)(Vp + ((size_t)bh * (S / 2) + j * 16) * DH)
                        + tid * 2;
      uint4* dV = (uint4*)&lV[buf][0];
      dV[tid * 2]     = vs[0];
      dV[tid * 2 + 1] = vs[1];
    }
#endif
    __syncthreads();

    // ---- logits: two 16x16 tiles (keys +0..15, +16..31), K-dim = dh = 64
    v8f c0 = {}, c1 = {};
    {
      v8u u;
#pragma unroll
      for (int v = 0; v < 8; ++v) u[v] = lK[buf][(8 * half + v) * 32 + lr];
      c0 = WMMA_BF16(aq0, __builtin_bit_cast(v16bf, u), c0);
#pragma unroll
      for (int v = 0; v < 8; ++v) u[v] = lK[buf][(16 + 8 * half + v) * 32 + lr];
      c0 = WMMA_BF16(aq1, __builtin_bit_cast(v16bf, u), c0);
#pragma unroll
      for (int v = 0; v < 8; ++v) u[v] = lK[buf][(8 * half + v) * 32 + 16 + lr];
      c1 = WMMA_BF16(aq0, __builtin_bit_cast(v16bf, u), c1);
#pragma unroll
      for (int v = 0; v < 8; ++v) u[v] = lK[buf][(16 + 8 * half + v) * 32 + 16 + lr];
      c1 = WMMA_BF16(aq1, __builtin_bit_cast(v16bf, u), c1);
    }

    // ---- online softmax (base-2); row v+8*half lives in this 16-lane half
    float sc[8];
#pragma unroll
    for (int v = 0; v < 8; ++v) {
      float tm = fmaxf(c0[v], c1[v]);
#pragma unroll
      for (int msk = 1; msk < 16; msk <<= 1)
        tm = fmaxf(tm, __shfl_xor(tm, msk, 32));
      const float mn = fmaxf(m[v], tm);
      sc[v] = exp2f(m[v] - mn);
      const float p0 = exp2f(c0[v] - mn);
      const float p1 = exp2f(c1[v] - mn);
      float rs = p0 + p1;
#pragma unroll
      for (int msk = 1; msk < 16; msk <<= 1)
        rs += __shfl_xor(rs, msk, 32);
      ssum[v] = ssum[v] * sc[v] + rs;
      m[v] = mn;
      const int M = v + 8 * half;
      lP[wave][M * 32 + lr]      = bf16bits(p0);
      lP[wave][M * 32 + 16 + lr] = bf16bits(p1);
    }
#pragma unroll
    for (int v = 0; v < 8; ++v) {
      o0[v] *= sc[v]; o1[v] *= sc[v]; o2[v] *= sc[v]; o3[v] *= sc[v];
    }
    __syncthreads();   // P written

    // ---- P back as 16x32 A fragment (transpose via LDS)
    v16bf aP;
    {
      v8u u;
      const unsigned* lPu = (const unsigned*)&lP[wave][0];
#pragma unroll
      for (int v = 0; v < 8; ++v) {
        const int kk = 16 * half + 2 * v;
        u[v] = lPu[lr * 16 + (kk >> 1)];
      }
      aP = __builtin_bit_cast(v16bf, u);
    }

    // ---- O += P @ V   (4 column tiles of dh)
#pragma unroll
    for (int t = 0; t < 4; ++t) {
      v8u u;
#pragma unroll
      for (int v = 0; v < 8; ++v)
        u[v] = lV[buf][(8 * half + v) * 64 + t * 16 + lr];
      v8f& o = (t == 0 ? o0 : t == 1 ? o1 : t == 2 ? o2 : o3);
      o = WMMA_BF16(aP, __builtin_bit_cast(v16bf, u), o);
    }
  }

  // ---- finalize: divide by softmax sum, write bf16 [b*S+s][h*64+dh]
#pragma unroll
  for (int v = 0; v < 8; ++v) {
    const float inv = 1.0f / ssum[v];
    const int r = b * S + q0 + v + 8 * half;
    const size_t base = (size_t)r * D + h * DH + lr;
    attnOut[base]      = bf16bits(o0[v] * inv);
    attnOut[base + 16] = bf16bits(o1[v] * inv);
    attnOut[base + 32] = bf16bits(o2[v] * inv);
    attnOut[base + 48] = bf16bits(o3[v] * inv);
  }
}

// ---------------------------------------------------------------------------
// Output projection: out = attn(bf16) @ Wo + bo, f32 result.
// A tile staged via TDM when available (data already bf16); the f32->bf16
// conversion of the W tile overlaps the DMA before the tensorcnt wait.
// ---------------------------------------------------------------------------
__global__ __launch_bounds__(128)
void out_proj_kernel(const unsigned* __restrict__ A,   // attn u32 [4096][512]
                     const float* __restrict__ W, const float* __restrict__ bias,
                     float* __restrict__ Y) {
  __shared__ unsigned lAu[64 * 16];
  __shared__ unsigned lBu[16 * 64];

  const int tid  = threadIdx.x;
  const int wave = tid >> 5;
  const int lane = tid & 31;
  const int half = lane >> 4;
  const int lr   = lane & 15;
  const int r0 = blockIdx.x * 64;
  const int c0 = blockIdx.y * 64;

  v8f acc0 = {}, acc1 = {}, acc2 = {}, acc3 = {};

  for (int k = 0; k < D; k += 32) {
    __syncthreads();
#ifdef HAVE_TDM
    if (wave == 0) {
      tdm_load_2d((unsigned)(size_t)(void*)lAu,
                  A + (size_t)r0 * (D / 2) + (k >> 1),
                  /*tile*/ 16, 64, /*tensor*/ D / 2, BB * S, /*stride*/ D / 2);
    }
#else
    {
      const int row = tid >> 1;
      const int o8  = (tid & 1) * 8;
      const uint4* src =
          (const uint4*)(A + (size_t)(r0 + row) * (D / 2) + (k >> 1) + o8);
      ((uint4*)lAu)[(row * 16 + o8) >> 2]       = src[0];
      ((uint4*)lAu)[((row * 16 + o8) >> 2) + 1] = src[1];
    }
#endif
    // B tile (needs f32->bf16 conversion, done in VALU while DMA runs)
    {
      const int kp = tid >> 3;
      const int cc = (tid & 7) * 8;
      const float* rp0 = W + (size_t)(k + 2 * kp) * D + c0 + cc;
      const float* rp1 = rp0 + D;
#pragma unroll
      for (int i = 0; i < 8; ++i)
        lBu[kp * 64 + cc + i] = pack_bf16(rp0[i], rp1[i]);
      if (k + 32 < D)
        __builtin_prefetch(W + (size_t)(k + 32 + 2 * kp) * D + c0 + cc, 0, 1);
    }
#ifdef HAVE_TDM
    if (wave == 0) __builtin_amdgcn_s_wait_tensorcnt(0);
#endif
    __syncthreads();

    v16bf a;
    {
      v8u u;
#pragma unroll
      for (int v = 0; v < 8; ++v) {
        const int kk = 16 * half + 2 * v;
        u[v] = lAu[(wave * 16 + lr) * 16 + (kk >> 1)];
      }
      a = __builtin_bit_cast(v16bf, u);
    }
#pragma unroll
    for (int t = 0; t < 4; ++t) {
      v8u u;
#pragma unroll
      for (int v = 0; v < 8; ++v)
        u[v] = lBu[(8 * half + v) * 64 + t * 16 + lr];
      v8f& acc = (t == 0 ? acc0 : t == 1 ? acc1 : t == 2 ? acc2 : acc3);
      acc = WMMA_BF16(a, __builtin_bit_cast(v16bf, u), acc);
    }
  }

  const int row0 = r0 + wave * 16;
#pragma unroll
  for (int t = 0; t < 4; ++t) {
    const v8f acc = (t == 0 ? acc0 : t == 1 ? acc1 : t == 2 ? acc2 : acc3);
    const int col = c0 + t * 16 + lr;
    const float bb = bias[col];
#pragma unroll
    for (int v = 0; v < 8; ++v) {
      const int r = row0 + v + 8 * half;
      Y[(size_t)r * D + col] = acc[v] + bb;
    }
  }
}

// ---------------------------------------------------------------------------
// Host-side launch.
// Inputs: query, key, value, Wq, bq, Wk, bk, Wv, bv, Wo, bo (all f32).
// Workspace: Qp (8MB) | Kp (8MB) | Vp (8MB) | attn bf16 (8MB) = 32MB.
// ---------------------------------------------------------------------------
extern "C" void kernel_launch(void* const* d_in, const int* in_sizes, int n_in,
                              void* d_out, int out_size, void* d_ws, size_t ws_size,
                              hipStream_t stream) {
  const float* query = (const float*)d_in[0];
  const float* key   = (const float*)d_in[1];
  const float* value = (const float*)d_in[2];
  const float* Wq    = (const float*)d_in[3];
  const float* bq    = (const float*)d_in[4];
  const float* Wk    = (const float*)d_in[5];
  const float* bk    = (const float*)d_in[6];
  const float* Wv    = (const float*)d_in[7];
  const float* bv    = (const float*)d_in[8];
  const float* Wo    = (const float*)d_in[9];
  const float* bo    = (const float*)d_in[10];
  float* out = (float*)d_out;

  char* ws = (char*)d_ws;
  unsigned*       Qp   = (unsigned*)(ws);                              // 8 MB
  unsigned*       Kp   = (unsigned*)(ws + (size_t)(8u << 20));         // 8 MB
  unsigned*       Vp   = (unsigned*)(ws + (size_t)(16u << 20));        // 8 MB
  unsigned short* attn = (unsigned short*)(ws + (size_t)(24u << 20));  // 8 MB

  const dim3 block(128, 1, 1);
  const dim3 gridP(BB * S / 64, D / 64, 1);   // 64 x 16 tiles of 64x64

  qkv_proj_kernel<<<gridP, block, 0, stream>>>(query, Wq, bq,
                                               (unsigned short*)Qp, nullptr, 0);
  qkv_proj_kernel<<<gridP, block, 0, stream>>>(key, Wk, bk, nullptr, Kp, 1);
  qkv_proj_kernel<<<gridP, block, 0, stream>>>(value, Wv, bv, nullptr, Vp, 2);

  const dim3 gridA(S / 64, H, BB);            // 32 x 16 x 2
  attn_kernel<<<gridA, block, 0, stream>>>(Qp, Kp, Vp, attn);

  out_proj_kernel<<<gridP, block, 0, stream>>>((const unsigned*)attn, Wo, bo, out);
}
